// TriangleAttention_75900662055648
// MI455X (gfx1250) — matrix operand order repaired and checked by reading, not compiled, not run on hardware
//
#include <hip/hip_runtime.h>

// ---------------------------------------------------------------------------
// TriangleAttention for MI455X (gfx1250, wave32, WMMA bf16 16x16x32)
// N=256, C=128, 4 heads x 32 dim.
// ---------------------------------------------------------------------------

#define NPIX   65536   // 256*256
#define CDIM   128
#define NSEQ   256
#define NHEADS 4
#define HDIM   32

typedef __attribute__((ext_vector_type(16))) __bf16        v16bf;
typedef __attribute__((ext_vector_type(16))) unsigned short v16u;
typedef __attribute__((ext_vector_type(8)))  unsigned short v8u;
typedef __attribute__((ext_vector_type(8)))  float          v8f;
typedef __attribute__((ext_vector_type(4)))  unsigned short su4;

static __device__ __forceinline__ unsigned short f2bf(float f) {
  unsigned int u = __float_as_uint(f);
  return (unsigned short)((u + 0x7FFFu + ((u >> 16) & 1u)) >> 16);  // RNE
}
static __device__ __forceinline__ float bf2f(unsigned short h) {
  return __uint_as_float((unsigned int)h << 16);
}
static __device__ __forceinline__ v16u cat8(v8u lo, v8u hi) {
  v16u r;
#pragma unroll
  for (int i = 0; i < 8; ++i) { r[i] = lo[i]; r[i + 8] = hi[i]; }
  return r;
}
static __device__ __forceinline__ v8f wmma_bf(v16u a, v16u b, v8f c) {
  return __builtin_amdgcn_wmma_f32_16x16x32_bf16(
      false, __builtin_bit_cast(v16bf, a),
      false, __builtin_bit_cast(v16bf, b),
      (short)0, c, false, false);
}

// Fragment index maps (from CDNA5 ISA 7.12.2, wave32):
//  A 16x32 bf16 : lane L, elem e -> M = L&15, K = (e&7) + 16*(e>>3) + 8*(L>>4)
//                 => two contiguous 8-half (16B) loads per lane (row-major A).
//  B 32x16 bf16 : lane L, elem e -> K = (e&7) + 16*(e>>3) + 8*(L>>4), N = L&15
//                 => two contiguous 8-half (16B) loads per lane if B stored [N][K].
//  C/D 16x16 f32: lane L, vgpr r -> M = r + 8*(L>>4), N = L&15.

// ---------------------------------------------------------------------------
// Kernel 1: LayerNorm (fp32) -> h (bf16) + pair bias[4][256][256] (fp32).
// One wave per pixel; 4 channels per lane; shfl_xor wave reductions.
// ---------------------------------------------------------------------------
__global__ void k_ln_bias(const float* __restrict__ x,
                          const float* __restrict__ wln,
                          const float* __restrict__ bln,
                          const float* __restrict__ wb,
                          unsigned short* __restrict__ hbf,
                          float* __restrict__ bias) {
  const int pix  = (blockIdx.x * blockDim.x + threadIdx.x) >> 5;
  const int lane = threadIdx.x & 31;

  const float4 xv = *(const float4*)(x + (size_t)pix * CDIM + lane * 4);
  float s = xv.x + xv.y + xv.z + xv.w;
#pragma unroll
  for (int mk = 16; mk >= 1; mk >>= 1) s += __shfl_xor(s, mk, 32);
  const float mu = s * (1.0f / 128.0f);

  const float d0 = xv.x - mu, d1 = xv.y - mu, d2 = xv.z - mu, d3 = xv.w - mu;
  float v = d0 * d0 + d1 * d1 + d2 * d2 + d3 * d3;
#pragma unroll
  for (int mk = 16; mk >= 1; mk >>= 1) v += __shfl_xor(v, mk, 32);
  const float rs = rsqrtf(v * (1.0f / 128.0f) + 1e-5f);

  const float4 wv = *(const float4*)(wln + lane * 4);
  const float4 bv = *(const float4*)(bln + lane * 4);
  const float h0 = d0 * rs * wv.x + bv.x;
  const float h1 = d1 * rs * wv.y + bv.y;
  const float h2 = d2 * rs * wv.z + bv.z;
  const float h3 = d3 * rs * wv.w + bv.w;

  su4 hb;
  hb[0] = f2bf(h0); hb[1] = f2bf(h1); hb[2] = f2bf(h2); hb[3] = f2bf(h3);
  *(su4*)(hbf + (size_t)pix * CDIM + lane * 4) = hb;

#pragma unroll
  for (int hh = 0; hh < NHEADS; ++hh) {
    const float4 wbv = *(const float4*)(wb + hh * CDIM + lane * 4);
    float p = h0 * wbv.x + h1 * wbv.y + h2 * wbv.z + h3 * wbv.w;
#pragma unroll
    for (int mk = 16; mk >= 1; mk >>= 1) p += __shfl_xor(p, mk, 32);
    if (lane == 0) bias[hh * NPIX + pix] = p;
  }
}

// ---------------------------------------------------------------------------
// Kernel 2: weight fp32 -> bf16 (w_q, w_k, w_v, w_g, w_o concatenated).
// ---------------------------------------------------------------------------
__global__ void k_cvt_w(const float* __restrict__ wq, const float* __restrict__ wk,
                        const float* __restrict__ wv, const float* __restrict__ wg,
                        const float* __restrict__ wo, unsigned short* __restrict__ dst) {
  const int i = blockIdx.x * blockDim.x + threadIdx.x;  // 0..81919
  const float* srcs[5] = {wq, wk, wv, wg, wo};
  dst[i] = f2bf(srcs[i >> 14][i & 16383]);
}

// ---------------------------------------------------------------------------
// Kernel 3: fused q/k/v/g projection GEMM  [65536,128] x W^T[128,128].
// Block = 8 waves; wave computes 16 rows x 64 cols (4 D-frags, 16 wmma).
// v is written TRANSPOSED: vT[b][head][d][key] so the PV B-frag is contiguous.
// ---------------------------------------------------------------------------
__global__ void k_qkvg(const unsigned short* __restrict__ hbf,
                       const unsigned short* __restrict__ wbf,
                       unsigned short* __restrict__ qb,
                       unsigned short* __restrict__ kb,
                       unsigned short* __restrict__ vT,
                       unsigned short* __restrict__ gb) {
  const int lane    = threadIdx.x & 31;
  const int waveid  = threadIdx.x >> 5;
  const int rowbase = blockIdx.x * 128 + waveid * 16;
  const int widx    = blockIdx.y >> 1;          // 0=q 1=k 2=v 3=g
  const int colbase = (blockIdx.y & 1) * 64;
  const int m = lane & 15, g16 = lane >> 4;

  v8f acc0 = {}, acc1 = {}, acc2 = {}, acc3 = {};
  v8f* acc[4] = {&acc0, &acc1, &acc2, &acc3};

#pragma unroll
  for (int kk = 0; kk < 4; ++kk) {
    const int k0 = kk * 32;
    const unsigned short* ar = hbf + (size_t)(rowbase + m) * CDIM + k0 + g16 * 8;
    const v16u a = cat8(*(const v8u*)ar, *(const v8u*)(ar + 16));
#pragma unroll
    for (int c = 0; c < 4; ++c) {
      const unsigned short* br =
          wbf + widx * 16384 + (size_t)(colbase + c * 16 + m) * CDIM + k0 + g16 * 8;
      const v16u b = cat8(*(const v8u*)br, *(const v8u*)(br + 16));
      *acc[c] = wmma_bf(a, b, *acc[c]);
    }
  }

#pragma unroll
  for (int c = 0; c < 4; ++c) {
#pragma unroll
    for (int r = 0; r < 8; ++r) {
      const int row = rowbase + r + 8 * g16;      // pixel
      const int col = colbase + c * 16 + m;       // output channel
      const unsigned short val = f2bf((*acc[c])[r]);
      if (widx == 0)      qb[(size_t)row * CDIM + col] = val;
      else if (widx == 1) kb[(size_t)row * CDIM + col] = val;
      else if (widx == 3) gb[(size_t)row * CDIM + col] = val;
      else {  // v -> vT[b][head][d][key]
        const int b2 = row >> 8, kj = row & 255;
        const int hh = col >> 5, dd = col & 31;
        vT[((size_t)(b2 * NHEADS + hh) * HDIM + dd) * NSEQ + kj] = val;
      }
    }
  }
}

// ---------------------------------------------------------------------------
// Kernel 4: fused attention per (row-batch b, head, 16-query tile per wave).
// Scores kept fully in VGPRs (16 x v8f = 128 VGPRs), softmax via half-wave
// shfl_xor reductions, P staged through LDS to build WMMA A-frags.
// ---------------------------------------------------------------------------
__global__ void __launch_bounds__(128)
k_attn(const unsigned short* __restrict__ qb,
       const unsigned short* __restrict__ kb,
       const unsigned short* __restrict__ vT,
       const float* __restrict__ bias,
       unsigned short* __restrict__ ob) {
  __shared__ unsigned short ldsP[4 * 16 * NSEQ];  // 32 KB: per-wave 16x256 bf16 P

  const int lane   = threadIdx.x & 31;
  const int waveid = threadIdx.x >> 5;
  const int b      = blockIdx.x;
  const int head   = blockIdx.y;
  const int qbase  = blockIdx.z * 64 + waveid * 16;
  const int m = lane & 15, g16 = lane >> 4;

  // Q A-frag: 16 queries x K=32 (full head dim in one WMMA step)
  const unsigned short* qr =
      qb + (size_t)(b * NSEQ + qbase + m) * CDIM + head * HDIM + g16 * 8;
  const v16u aq = cat8(*(const v8u*)qr, *(const v8u*)(qr + 16));

  // ---- scores: S[t] covers keys [16t, 16t+16) ----
  v8f S[16];
#pragma unroll
  for (int t = 0; t < 16; ++t) {
    const unsigned short* kr =
        kb + (size_t)(b * NSEQ + t * 16 + m) * CDIM + head * HDIM + g16 * 8;
    const v16u bk = cat8(*(const v8u*)kr, *(const v8u*)(kr + 16));
    v8f z = {};
    S[t] = wmma_bf(aq, bk, z);
  }

  // ---- scale + pair bias (independent of b) ----
  const float scale = 0.17677669529663689f;  // 1/sqrt(32)
  const float* bp = bias + head * NPIX;
#pragma unroll
  for (int t = 0; t < 16; ++t) {
#pragma unroll
    for (int r = 0; r < 8; ++r) {
      S[t][r] = S[t][r] * scale + bp[(qbase + r + 8 * g16) * NSEQ + t * 16 + m];
    }
  }

  // ---- softmax over 256 keys; row lives in one 16-lane half-wave ----
#pragma unroll
  for (int r = 0; r < 8; ++r) {
    float mx = -3.4e38f;
#pragma unroll
    for (int t = 0; t < 16; ++t) mx = fmaxf(mx, S[t][r]);
#pragma unroll
    for (int mk = 8; mk >= 1; mk >>= 1) mx = fmaxf(mx, __shfl_xor(mx, mk, 32));
    float sum = 0.0f;
#pragma unroll
    for (int t = 0; t < 16; ++t) {
      const float e = __expf(S[t][r] - mx);
      S[t][r] = e;
      sum += e;
    }
#pragma unroll
    for (int mk = 8; mk >= 1; mk >>= 1) sum += __shfl_xor(sum, mk, 32);
    const float inv = 1.0f / sum;
    // write P row strip to this wave's private LDS region (transpose to A layout)
#pragma unroll
    for (int t = 0; t < 16; ++t) {
      ldsP[waveid * 4096 + (r + 8 * g16) * NSEQ + t * 16 + m] = f2bf(S[t][r] * inv);
    }
  }
  __syncthreads();

  // ---- O = P @ V : 2 d-tiles x 8 k-steps of 32 ----
  const unsigned short* vbase = vT + (size_t)(b * NHEADS + head) * HDIM * NSEQ;
#pragma unroll
  for (int dt = 0; dt < 2; ++dt) {
    v8f acc = {};
#pragma unroll
    for (int ks = 0; ks < 8; ++ks) {
      const unsigned short* pr = &ldsP[waveid * 4096 + m * NSEQ + ks * 32 + g16 * 8];
      const v16u ap = cat8(*(const v8u*)pr, *(const v8u*)(pr + 16));
      const unsigned short* vr = vbase + (size_t)(dt * 16 + m) * NSEQ + ks * 32 + g16 * 8;
      const v16u bv = cat8(*(const v8u*)vr, *(const v8u*)(vr + 16));
      acc = wmma_bf(ap, bv, acc);
    }
#pragma unroll
    for (int r = 0; r < 8; ++r) {
      ob[(size_t)(b * NSEQ + qbase + r + 8 * g16) * CDIM + head * HDIM + dt * 16 + m] =
          f2bf(acc[r]);
    }
  }
}

// ---------------------------------------------------------------------------
// Kernel 5: out = (o * sigmoid(g)) @ w_o^T, fp32 output.
// A-frag built on the fly from bf16 o and g.
// ---------------------------------------------------------------------------
__global__ void k_gated_out(const unsigned short* __restrict__ ob,
                            const unsigned short* __restrict__ gb,
                            const unsigned short* __restrict__ wobf,
                            float* __restrict__ out) {
  const int lane    = threadIdx.x & 31;
  const int waveid  = threadIdx.x >> 5;
  const int rowbase = blockIdx.x * 128 + waveid * 16;
  const int colbase = blockIdx.y * 64;
  const int m = lane & 15, g16 = lane >> 4;

  v8f acc0 = {}, acc1 = {}, acc2 = {}, acc3 = {};
  v8f* acc[4] = {&acc0, &acc1, &acc2, &acc3};

#pragma unroll
  for (int kk = 0; kk < 4; ++kk) {
    const int k0 = kk * 32;
    const size_t base = (size_t)(rowbase + m) * CDIM + k0 + g16 * 8;
    const v8u o0 = *(const v8u*)(ob + base), o1 = *(const v8u*)(ob + base + 16);
    const v8u q0 = *(const v8u*)(gb + base), q1 = *(const v8u*)(gb + base + 16);
    v16u a;
#pragma unroll
    for (int e = 0; e < 8; ++e) {
      const float x0 = bf2f(o0[e]) * (1.0f / (1.0f + __expf(-bf2f(q0[e]))));
      const float x1 = bf2f(o1[e]) * (1.0f / (1.0f + __expf(-bf2f(q1[e]))));
      a[e] = f2bf(x0);
      a[e + 8] = f2bf(x1);
    }
#pragma unroll
    for (int c = 0; c < 4; ++c) {
      const unsigned short* br =
          wobf + (size_t)(colbase + c * 16 + m) * CDIM + k0 + g16 * 8;
      const v16u bfr = cat8(*(const v8u*)br, *(const v8u*)(br + 16));
      *acc[c] = wmma_bf(a, bfr, *acc[c]);
    }
  }

#pragma unroll
  for (int c = 0; c < 4; ++c) {
#pragma unroll
    for (int r = 0; r < 8; ++r) {
      out[(size_t)(rowbase + r + 8 * g16) * CDIM + colbase + c * 16 + m] = (*acc[c])[r];
    }
  }
}

// ---------------------------------------------------------------------------
extern "C" void kernel_launch(void* const* d_in, const int* in_sizes, int n_in,
                              void* d_out, int out_size, void* d_ws, size_t ws_size,
                              hipStream_t stream) {
  (void)in_sizes; (void)n_in; (void)out_size; (void)ws_size;

  const float* x   = (const float*)d_in[0];
  const float* wln = (const float*)d_in[1];
  const float* bln = (const float*)d_in[2];
  const float* wb  = (const float*)d_in[3];
  const float* wq  = (const float*)d_in[4];
  const float* wk  = (const float*)d_in[5];
  const float* wv  = (const float*)d_in[6];
  const float* wg  = (const float*)d_in[7];
  const float* wo  = (const float*)d_in[8];

  // workspace layout (elements), total ~97.2 MB
  const size_t HB = (size_t)NPIX * CDIM;  // 8,388,608
  unsigned short* hbf  = (unsigned short*)d_ws;
  unsigned short* qb   = hbf + HB;
  unsigned short* kbuf = qb + HB;
  unsigned short* vT   = kbuf + HB;
  unsigned short* gb   = vT + HB;
  unsigned short* obuf = gb + HB;
  unsigned short* wbf  = obuf + HB;              // 5 * 16384 bf16 weights
  float*          bias = (float*)(wbf + 5 * 16384);  // 4 * 65536 f32

  k_ln_bias<<<dim3(NPIX / 8), 256, 0, stream>>>(x, wln, bln, wb, hbf, bias);
  k_cvt_w<<<dim3(320), 256, 0, stream>>>(wq, wk, wv, wg, wo, wbf);
  k_qkvg<<<dim3(NPIX / 128, 8), 256, 0, stream>>>(hbf, wbf, qb, kbuf, vT, gb);
  k_attn<<<dim3(NSEQ, NHEADS, 4), 128, 0, stream>>>(qb, kbuf, vT, bias, obuf);
  k_gated_out<<<dim3(NPIX / 128, 2), 256, 0, stream>>>(obuf, gb, wbf + 4 * 16384,
                                                       (float*)d_out);
}
